// VideoObjectSegmentationModel_3745211482283
// MI455X (gfx1250) — matrix operand order repaired and checked by reading, not compiled, not run on hardware
//
#include <hip/hip_runtime.h>
#include <hip/hip_bf16.h>

// ---------------------------------------------------------------------------
// Video object segmentation model, MI455X (gfx1250), wave32 + WMMA f16.
//   prep(weights->f16, reorder/pad) ->
//   im2col1 -> WMMA gemm (conv1+relu, NHWC f16) ->
//   im2col2 -> WMMA gemm (conv2+relu) ->
//   im2col3 -> WMMA gemm (conv3+relu) ->
//   WMMA gemm fc (+relu, f32+f16 out) -> heads (ot/ct, VALU) ->
//   WMMA gemm m1 (+bias) ->
//   up 21->42 -> fused WMMA conv c1 + residual relu ->
//   up 42->84 -> fused WMMA conv c2 + residual relu + c3 + sigmoid
//                + per-object flow + ct + bilinear warp -> out
// GEMMs use 32x32 macro-tiles per wave (2x2 WMMA register blocking).
// dec2 stages u84 rows into LDS with GLOBAL_LOAD_ASYNC_TO_LDS (if available).
// ---------------------------------------------------------------------------

typedef _Float16 half16v __attribute__((ext_vector_type(16)));
typedef _Float16 half8v  __attribute__((ext_vector_type(8)));
typedef float    f32x8   __attribute__((ext_vector_type(8)));
typedef int      v4i     __attribute__((vector_size(16)));

static constexpr int BB = 512, EMB = 512;
static constexpr int FCS = 7 * 7 * 64;

#if defined(__AMDGCN__) && __has_builtin(__builtin_amdgcn_global_load_async_to_lds_b128) && \
    __has_builtin(__builtin_amdgcn_s_wait_asynccnt)
#define HAVE_ASYNC_LDS 1
#else
#define HAVE_ASYNC_LDS 0
#endif

// ---------------- WMMA fragment loaders (16x16x32 f16, wave32) -------------
__device__ __forceinline__ half16v load_a_frag(const _Float16* __restrict__ Ab, int lda) {
  int l = threadIdx.x & 31;
  int m = l & 15;
  int sel = l >> 4;
  const _Float16* p0 = Ab + m * lda + sel * 8;      // K = sel*8 .. +7
  half8v lo = *(const half8v*)p0;
  half8v hi = *(const half8v*)(p0 + 16);            // K = 16+sel*8 .. +7
  return __builtin_shufflevector(lo, hi, 0, 1, 2, 3, 4, 5, 6, 7,
                                 8, 9, 10, 11, 12, 13, 14, 15);
}

__device__ __forceinline__ half16v load_b_frag(const _Float16* __restrict__ Bb, int ldb) {
  int l = threadIdx.x & 31;
  int n = l & 15;
  int sel = l >> 4;
  const _Float16* p0 = Bb + n * ldb + sel * 8;      // K = sel*8 .. +7
  half8v lo = *(const half8v*)p0;
  half8v hi = *(const half8v*)(p0 + 16);            // K = 16+sel*8 .. +7
  return __builtin_shufflevector(lo, hi, 0, 1, 2, 3, 4, 5, 6, 7,
                                 8, 9, 10, 11, 12, 13, 14, 15);
}

__device__ __forceinline__ f32x8 wmma_f32_f16(half16v a, half16v b, f32x8 c) {
  return __builtin_amdgcn_wmma_f32_16x16x32_f16(false, a, false, b, (short)0, c,
                                                false, false);
}

// ---------------- generic WMMA GEMM: C = act(A * B^T + bias) ---------------
// A: [M][K] f16 row-major, B: [N][K] f16 row-major.
// M, N multiples of 32; K multiple of 32. One 32x32 macro-tile per wave.
template <bool RELU, bool BIAS, bool SF32, bool SF16>
__global__ __launch_bounds__(256) void wmma_gemm_k(
    const _Float16* __restrict__ A, const _Float16* __restrict__ B,
    const float* __restrict__ bias, float* __restrict__ Cf,
    _Float16* __restrict__ Ch, int M, int N, int K, int Nreal) {
  int wave = blockIdx.x * (blockDim.x >> 5) + (threadIdx.x >> 5);
  int stride = gridDim.x * (blockDim.x >> 5);
  int ntiles = N >> 5;
  int total = (M >> 5) * ntiles;
  int l = threadIdx.x & 31;
  int sel = l >> 4;
  for (int t = wave; t < total; t += stride) {
    int mt = t / ntiles;
    int nt = t - mt * ntiles;
    const _Float16* Ab = A + (size_t)mt * 32 * K;
    const _Float16* Bb = B + (size_t)nt * 32 * K;
    const _Float16* Ab1 = Ab + (size_t)16 * K;
    const _Float16* Bb1 = Bb + (size_t)16 * K;
    f32x8 a00 = {0.f, 0.f, 0.f, 0.f, 0.f, 0.f, 0.f, 0.f};
    f32x8 a01 = a00, a10 = a00, a11 = a00;
    for (int kk = 0; kk < K; kk += 32) {
      if (kk + 128 < K) {
        __builtin_prefetch(Ab + kk + 128, 0, 3);
        __builtin_prefetch(Bb + kk + 128, 0, 3);
      }
      half16v af0 = load_a_frag(Ab + kk, K);
      half16v af1 = load_a_frag(Ab1 + kk, K);
      half16v bf0 = load_b_frag(Bb + kk, K);
      half16v bf1 = load_b_frag(Bb1 + kk, K);
      a00 = wmma_f32_f16(af0, bf0, a00);
      a01 = wmma_f32_f16(af0, bf1, a01);
      a10 = wmma_f32_f16(af1, bf0, a10);
      a11 = wmma_f32_f16(af1, bf1, a11);
    }
    auto epi = [&](const f32x8& acc, int ti, int tj) {
      int n = nt * 32 + tj * 16 + (l & 15);
      if (n < Nreal) {
        float bv = BIAS ? bias[n] : 0.f;
#pragma unroll
        for (int i = 0; i < 8; ++i) {
          int m = mt * 32 + ti * 16 + sel * 8 + i;
          float v = acc[i] + bv;
          if (RELU) v = fmaxf(v, 0.f);
          size_t o = (size_t)m * Nreal + n;
          if (SF32) Cf[o] = v;
          if (SF16) Ch[o] = (_Float16)v;
        }
      }
    };
    epi(a00, 0, 0);
    epi(a01, 0, 1);
    epi(a10, 1, 0);
    epi(a11, 1, 1);
  }
}

// ---------------- weight prep: f32 -> f16, reorder / zero-pad ---------------
static constexpr int N_FCW = EMB * FCS;             // reorder k -> NHWC
static constexpr int N_M1W = 10592 * EMB;           // pad rows 10584..10591
static constexpr int N_CW1 = 32 * 128;
static constexpr int N_CW2 = 64 * 512;
static constexpr int N_CW3 = 64 * 576;
static constexpr int N_CCW = 32 * 224;              // c1w / c2w padded [32][224]
static constexpr int PREP_TOTAL = N_FCW + N_M1W + N_CW1 + N_CW2 + N_CW3 + 2 * N_CCW;

__global__ void prep_weights_k(
    const float* __restrict__ fcw, const float* __restrict__ m1w,
    const float* __restrict__ cw1, const float* __restrict__ cw2,
    const float* __restrict__ cw3, const float* __restrict__ c1w,
    const float* __restrict__ c2w, _Float16* __restrict__ fcw16,
    _Float16* __restrict__ m1w16, _Float16* __restrict__ cw1_16,
    _Float16* __restrict__ cw2_16, _Float16* __restrict__ cw3_16,
    _Float16* __restrict__ c1w16, _Float16* __restrict__ c2w16) {
  int idx = blockIdx.x * blockDim.x + threadIdx.x;
  if (idx >= PREP_TOTAL) return;
  int i = idx;
  if (i < N_FCW) {  // fcw[n][c*49+s] -> fcw16[n][s*64+c]
    int n = i / FCS, kp = i - n * FCS;
    int s = kp >> 6, c = kp & 63;
    fcw16[i] = (_Float16)fcw[(size_t)n * FCS + c * 49 + s];
    return;
  }
  i -= N_FCW;
  if (i < N_M1W) {
    int n = i >> 9, k = i & 511;
    m1w16[i] = (n < 10584) ? (_Float16)m1w[(size_t)n * 512 + k] : (_Float16)0.f;
    return;
  }
  i -= N_M1W;
  if (i < N_CW1) { cw1_16[i] = (_Float16)cw1[i]; return; }
  i -= N_CW1;
  if (i < N_CW2) { cw2_16[i] = (_Float16)cw2[i]; return; }
  i -= N_CW2;
  if (i < N_CW3) { cw3_16[i] = (_Float16)cw3[i]; return; }
  i -= N_CW3;
  if (i < N_CCW) {
    int n = i / 224, k = i - n * 224;
    c1w16[i] = (n < 24 && k < 216) ? (_Float16)c1w[n * 216 + k] : (_Float16)0.f;
    return;
  }
  i -= N_CCW;
  {
    int n = i / 224, k = i - n * 224;
    c2w16[i] = (n < 24 && k < 216) ? (_Float16)c2w[n * 216 + k] : (_Float16)0.f;
  }
}

// ---------------- im2col gathers (encoder) ---------------------------------
__global__ void im2col1_k(const float* __restrict__ inp, _Float16* __restrict__ A) {
  int idx = blockIdx.x * blockDim.x + threadIdx.x;
  if (idx >= 204800 * 128) return;
  int k = idx & 127, m = idx >> 7;
  int b = m / 400, r = m - b * 400, oy = r / 20, ox = r - (r / 20) * 20;
  int c = k >> 6, rr = k & 63, ki = rr >> 3, kj = rr & 7;
  A[idx] = (_Float16)inp[(((size_t)b * 2 + c) * 84 + oy * 4 + ki) * 84 + ox * 4 + kj];
}

__global__ void im2col2_k(const _Float16* __restrict__ h1, _Float16* __restrict__ A) {
  int idx = blockIdx.x * blockDim.x + threadIdx.x;
  if (idx >= 41472 * 512) return;
  int k = idx & 511, m = idx >> 9;
  int b = m / 81, r = m - b * 81, oy = r / 9, ox = r - (r / 9) * 9;
  int c = k >> 4, rr = k & 15, ki = rr >> 2, kj = rr & 3;
  A[idx] = h1[(((size_t)b * 20 + oy * 2 + ki) * 20 + ox * 2 + kj) * 32 + c];
}

__global__ void im2col3_k(const _Float16* __restrict__ h2, _Float16* __restrict__ A) {
  int idx = blockIdx.x * blockDim.x + threadIdx.x;
  if (idx >= 25088 * 576) return;
  int m = idx / 576, k = idx - m * 576;
  int b = m / 49, r = m - b * 49, oy = r / 7, ox = r - (r / 7) * 7;
  int c = k / 9, rr = k - c * 9, ki = rr / 3, kj = rr - (rr / 3) * 3;
  A[idx] = h2[(((size_t)b * 9 + oy + ki) * 9 + ox + kj) * 64 + c];
}

// ---------------- heads: ot = x@otw^T+otb, ct = x@ctw^T+ctb ----------------
__global__ void heads_k(const float* __restrict__ x, const float* __restrict__ otw,
                        const float* __restrict__ otb, const float* __restrict__ ctw,
                        const float* __restrict__ ctb, float* __restrict__ ot,
                        float* __restrict__ ct) {
  int idx = blockIdx.x * blockDim.x + threadIdx.x;
  if (idx >= BB * 42) return;
  int b = idx / 42, j = idx - b * 42;
  const float* xb = x + (size_t)b * EMB;
  const float* wr = (j < 40) ? (otw + (size_t)j * EMB) : (ctw + (size_t)(j - 40) * EMB);
  float s = (j < 40) ? otb[j] : ctb[j - 40];
  for (int e = 0; e < EMB; ++e) s += xb[e] * wr[e];
  if (j < 40) ot[b * 40 + j] = s;
  else        ct[b * 2 + (j - 40)] = s;
}

// ---------------- bilinear upsample (align_corners) ------------------------
// m21: [b][c][21][21] f32 -> m42u: [b][y][c][x] f16 (42x42)
__global__ void up21_42_k(const float* __restrict__ m21, _Float16* __restrict__ dst) {
  int idx = blockIdx.x * blockDim.x + threadIdx.x;
  if (idx >= BB * 42 * 24 * 42) return;
  int x = idx % 42;
  int t1 = idx / 42;
  int c = t1 % 24;
  int t2 = t1 / 24;
  int y = t2 % 42;
  int b = t2 / 42;
  const float s = 20.f / 41.f;
  float py = y * s; int ly = (int)py; if (ly > 19) ly = 19; float wy = py - ly;
  float px = x * s; int lx = (int)px; if (lx > 19) lx = 19; float wx = px - lx;
  const float* mb = m21 + ((size_t)b * 24 + c) * 441;
  float v00 = mb[ly * 21 + lx],       v01 = mb[ly * 21 + lx + 1];
  float v10 = mb[(ly + 1) * 21 + lx], v11 = mb[(ly + 1) * 21 + lx + 1];
  dst[idx] = (_Float16)((1.f - wy) * ((1.f - wx) * v00 + wx * v01) +
                        wy * ((1.f - wx) * v10 + wx * v11));
}

// t_u: [b][y][c][x] f16 (42x42) -> u84: [b][y][c][x] f16 (84x84)
__global__ void up42_84_k(const _Float16* __restrict__ t_u, _Float16* __restrict__ dst) {
  int idx = blockIdx.x * blockDim.x + threadIdx.x;
  if (idx >= BB * 84 * 24 * 84) return;
  int x = idx % 84;
  int t1 = idx / 84;
  int c = t1 % 24;
  int t2 = t1 / 24;
  int y = t2 % 84;
  int b = t2 / 84;
  const float s = 41.f / 83.f;
  float py = y * s; int ly = (int)py; if (ly > 40) ly = 40; float wy = py - ly;
  float px = x * s; int lx = (int)px; if (lx > 40) lx = 40; float wx = px - lx;
  size_t r0 = ((size_t)(b * 42 + ly) * 24 + c) * 42;
  size_t r1 = r0 + 24 * 42;
  float v00 = (float)t_u[r0 + lx], v01 = (float)t_u[r0 + lx + 1];
  float v10 = (float)t_u[r1 + lx], v11 = (float)t_u[r1 + lx + 1];
  dst[idx] = (_Float16)((1.f - wy) * ((1.f - wx) * v00 + wx * v01) +
                        wy * ((1.f - wx) * v10 + wx * v11));
}

// ------------- decoder stage 1: 3x3 conv (WMMA) + residual relu ------------
// in/out layout [b][y][c][x] f16, 42x42, 24 ch; W: [32][224] f16 padded.
// im2col lda = 216 (k-pads only meet zero weights; guard tail zeroed).
__global__ __launch_bounds__(256) void dec1_conv_k(
    const _Float16* __restrict__ m42u, const _Float16* __restrict__ Wc1,
    const float* __restrict__ c1b, _Float16* __restrict__ t_u) {
  __shared__ alignas(32) _Float16 imcol[48 * 216 + 16];
  int b = blockIdx.x;
  int tid = threadIdx.x;
  int w = tid >> 5;
  int l = tid & 31;
  int sel = l >> 4;
  if (tid < 16) imcol[48 * 216 + tid] = (_Float16)0.f;
  for (int y = 0; y < 42; ++y) {
    for (int i = tid; i < 48 * 216; i += 256) {
      int px = i / 216, k = i - px * 216;
      _Float16 v = (_Float16)0.f;
      if (px < 42) {
        int c = k / 9, r = k - c * 9;
        int dy = r / 3 - 1, dx = r - (r / 3) * 3 - 1;
        int yy = y + dy, xx = px + dx;
        if ((unsigned)yy < 42u && (unsigned)xx < 42u)
          v = m42u[((size_t)(b * 42 + yy) * 24 + c) * 42 + xx];
      }
      imcol[i] = v;
    }
    __syncthreads();
    if (w < 6) {  // 3 M-tiles x 2 N-tiles
      int mt = w >> 1, nt = w & 1;
      f32x8 acc = {0.f, 0.f, 0.f, 0.f, 0.f, 0.f, 0.f, 0.f};
      for (int kk = 0; kk < 224; kk += 32) {
        half16v af = load_a_frag(imcol + mt * 16 * 216 + kk, 216);
        half16v bf = load_b_frag(Wc1 + nt * 16 * 224 + kk, 224);
        acc = wmma_f32_f16(af, bf, acc);
      }
      int c = nt * 16 + (l & 15);
      if (c < 24) {
        float bv = c1b[c];
#pragma unroll
        for (int i = 0; i < 8; ++i) {
          int px = mt * 16 + sel * 8 + i;
          if (px < 42) {
            size_t o = ((size_t)(b * 42 + y) * 24 + c) * 42 + px;
            float res = (float)m42u[o] + acc[i] + bv;
            t_u[o] = (_Float16)fmaxf(res, 0.f);
          }
        }
      }
    }
    __syncthreads();
  }
}

// ------------- decoder stage 2 (fully fused tail) --------------------------
// per image: 3x3 conv c2 (WMMA) + residual relu -> c3 1x1 + sigmoid ->
// per-object flow + ct -> coords -> bilinear warp of inp[:,1] -> out.
// u84 channel-rows staged through LDS ring via async global->LDS copies.
__global__ __launch_bounds__(256) void dec2_fused_k(
    const _Float16* __restrict__ u84, const _Float16* __restrict__ Wc2,
    const float* __restrict__ c2b, const float* __restrict__ c3w,
    const float* __restrict__ c3b, const float* __restrict__ ot,
    const float* __restrict__ ct, const float* __restrict__ inp,
    float* __restrict__ out) {
  __shared__ alignas(32) _Float16 imcol[96 * 216 + 16];
  __shared__ float vbuf[24][84];
  __shared__ float ot_s[40], ct_s[2], c3w_s[20 * 24], c3b_s[20];
#if HAVE_ASYNC_LDS
  __shared__ alignas(16) _Float16 urow[3][24 * 84];  // ring of u84 rows
#endif
  int b = blockIdx.x;
  int tid = threadIdx.x;
  int w = tid >> 5;
  int l = tid & 31;
  int sel = l >> 4;
  if (tid < 40) ot_s[tid] = ot[b * 40 + tid];
  if (tid < 2) ct_s[tid] = ct[b * 2 + tid];
  for (int i = tid; i < 20 * 24; i += 256) c3w_s[i] = c3w[i];
  if (tid < 20) c3b_s[tid] = c3b[tid];
  if (tid < 16) imcol[96 * 216 + tid] = (_Float16)0.f;
  const float* src = inp + ((size_t)b * 2 + 1) * 84 * 84;
#if HAVE_ASYNC_LDS
  auto issue_row = [&](int yy) {  // 24*84 f16 = 4032B = 252 x 16B chunks
    if (tid < 252) {
      const _Float16* gp = u84 + ((size_t)(b * 84 + yy) * 24) * 84 + tid * 8;
      __builtin_amdgcn_global_load_async_to_lds_b128(
          (__attribute__((address_space(1))) v4i*)gp,
          (__attribute__((address_space(3))) v4i*)(&urow[yy % 3][tid * 8]), 0, 0);
    }
  };
  issue_row(0);
#endif
  __syncthreads();
  for (int y = 0; y < 84; ++y) {
#if HAVE_ASYNC_LDS
    if (y + 1 < 84) issue_row(y + 1);
    __builtin_amdgcn_s_wait_asynccnt(0);
    __syncthreads();  // all waves' row parts visible
#endif
    for (int i = tid; i < 96 * 216; i += 256) {
      int px = i / 216, k = i - px * 216;
      _Float16 v = (_Float16)0.f;
      if (px < 84) {
        int c = k / 9, r = k - c * 9;
        int dy = r / 3 - 1, dx = r - (r / 3) * 3 - 1;
        int yy = y + dy, xx = px + dx;
        if ((unsigned)yy < 84u && (unsigned)xx < 84u)
#if HAVE_ASYNC_LDS
          v = urow[yy % 3][c * 84 + xx];
#else
          v = u84[((size_t)(b * 84 + yy) * 24 + c) * 84 + xx];
#endif
      }
      imcol[i] = v;
    }
    __syncthreads();
    for (int t = w; t < 12; t += 8) {  // 6 M-tiles x 2 N-tiles
      int mt = t % 6, nt = t / 6;
      f32x8 acc = {0.f, 0.f, 0.f, 0.f, 0.f, 0.f, 0.f, 0.f};
      for (int kk = 0; kk < 224; kk += 32) {
        half16v af = load_a_frag(imcol + mt * 16 * 216 + kk, 216);
        half16v bf = load_b_frag(Wc2 + nt * 16 * 224 + kk, 224);
        acc = wmma_f32_f16(af, bf, acc);
      }
      int c = nt * 16 + (l & 15);
      if (c < 24) {
        float bv = c2b[c];
#pragma unroll
        for (int i = 0; i < 8; ++i) {
          int px = mt * 16 + sel * 8 + i;
          if (px < 84) {
#if HAVE_ASYNC_LDS
            float u = (float)urow[y % 3][c * 84 + px];
#else
            float u = (float)u84[((size_t)(b * 84 + y) * 24 + c) * 84 + px];
#endif
            vbuf[c][px] = fmaxf(u + acc[i] + bv, 0.f);
          }
        }
      }
    }
    __syncthreads();
    for (int x = tid; x < 84; x += 256) {
      float f0 = ct_s[0], f1 = ct_s[1];
      for (int k = 0; k < 20; ++k) {
        float s = c3b_s[k];
        for (int c = 0; c < 24; ++c) s += c3w_s[k * 24 + c] * vbuf[c][x];
        float mk = 1.f / (1.f + __expf(-s));
        f0 += mk * ot_s[2 * k];
        f1 += mk * ot_s[2 * k + 1];
      }
      float ys = 0.84f * f0 + (float)y;   // FLOW_C * H = 0.84
      float xs = 0.84f * f1 + (float)x;   // FLOW_C * W = 0.84
      int x0 = (int)floorf(xs); x0 = x0 < 0 ? 0 : (x0 > 83 ? 83 : x0);
      int x1 = x0 + 1 > 83 ? 83 : x0 + 1;
      int y0 = (int)floorf(ys); y0 = y0 < 0 ? 0 : (y0 > 83 ? 83 : y0);
      int y1 = y0 + 1 > 83 ? 83 : y0 + 1;
      float xc = fminf(fmaxf(xs, 0.f), 83.f);
      float yc = fminf(fmaxf(ys, 0.f), 83.f);
      float Ia = src[y0 * 84 + x0], Ibv = src[y1 * 84 + x0];
      float Ic = src[y0 * 84 + x1], Id = src[y1 * 84 + x1];
      float wx1 = (float)x1 - xc, wx0 = xc - (float)x0;
      float wy1 = (float)y1 - yc, wy0 = yc - (float)y0;
      out[((size_t)b * 84 + y) * 84 + x] =
          wx1 * wy1 * Ia + wx1 * wy0 * Ibv + wx0 * wy1 * Ic + wx0 * wy0 * Id;
    }
    __syncthreads();
  }
}

// ---------------------------------------------------------------------------
extern "C" void kernel_launch(void* const* d_in, const int* in_sizes, int n_in,
                              void* d_out, int out_size, void* d_ws, size_t ws_size,
                              hipStream_t stream) {
  const float* inp = (const float*)d_in[0];
  const float* cw1 = (const float*)d_in[1];
  const float* cb1 = (const float*)d_in[2];
  const float* cw2 = (const float*)d_in[3];
  const float* cb2 = (const float*)d_in[4];
  const float* cw3 = (const float*)d_in[5];
  const float* cb3 = (const float*)d_in[6];
  const float* fcw = (const float*)d_in[7];
  const float* fcb = (const float*)d_in[8];
  const float* otw = (const float*)d_in[9];
  const float* otb = (const float*)d_in[10];
  const float* ctw = (const float*)d_in[11];
  const float* ctb = (const float*)d_in[12];
  const float* m1w = (const float*)d_in[13];
  const float* m1b = (const float*)d_in[14];
  const float* c1w = (const float*)d_in[15];
  const float* c1b = (const float*)d_in[16];
  const float* c2w = (const float*)d_in[17];
  const float* c2b = (const float*)d_in[18];
  const float* c3w = (const float*)d_in[19];
  const float* c3b = (const float*)d_in[20];
  float* outp = (float*)d_out;

  char* ws = (char*)d_ws;
  size_t off = 0;
  auto alloc = [&](size_t bytes) {
    size_t r = off;
    off += (bytes + 255) & ~(size_t)255;
    return r;
  };
  _Float16* imcolA = (_Float16*)(ws + alloc(204800ull * 128 * 2));  // reused x3
  _Float16* h1     = (_Float16*)(ws + alloc(204800ull * 32 * 2));
  _Float16* h2     = (_Float16*)(ws + alloc(41472ull * 64 * 2));
  _Float16* h3     = (_Float16*)(ws + alloc(512ull * 3136 * 2));
  float*    x32    = (float*)(ws + alloc(512ull * 512 * 4));
  _Float16* x16    = (_Float16*)(ws + alloc(512ull * 512 * 2));
  float*    otB    = (float*)(ws + alloc(512ull * 40 * 4));
  float*    ctB    = (float*)(ws + alloc(512ull * 2 * 4));
  _Float16* fcw16  = (_Float16*)(ws + alloc((size_t)N_FCW * 2));
  _Float16* m1w16  = (_Float16*)(ws + alloc((size_t)N_M1W * 2));
  _Float16* cw1_16 = (_Float16*)(ws + alloc((size_t)N_CW1 * 2));
  _Float16* cw2_16 = (_Float16*)(ws + alloc((size_t)N_CW2 * 2));
  _Float16* cw3_16 = (_Float16*)(ws + alloc((size_t)N_CW3 * 2));
  _Float16* c1w16  = (_Float16*)(ws + alloc((size_t)N_CCW * 2));
  _Float16* c2w16  = (_Float16*)(ws + alloc((size_t)N_CCW * 2));
  float*    m21    = (float*)(ws + alloc(512ull * 10584 * 4));
  _Float16* m42u   = (_Float16*)(ws + alloc(512ull * 42 * 24 * 42 * 2));
  _Float16* t_u    = (_Float16*)(ws + alloc(512ull * 42 * 24 * 42 * 2));
  _Float16* u84    = (_Float16*)(ws + alloc(512ull * 84 * 24 * 84 * 2));
  (void)ws_size; (void)in_sizes; (void)n_in; (void)out_size;

  // 0) weight prep
  prep_weights_k<<<(PREP_TOTAL + 255) / 256, 256, 0, stream>>>(
      fcw, m1w, cw1, cw2, cw3, c1w, c2w,
      fcw16, m1w16, cw1_16, cw2_16, cw3_16, c1w16, c2w16);

  // 1) conv1: im2col + GEMM  (M=204800, N=32, K=128)
  im2col1_k<<<(204800 * 128 + 255) / 256, 256, 0, stream>>>(inp, imcolA);
  wmma_gemm_k<true, true, false, true><<<800, 256, 0, stream>>>(
      imcolA, cw1_16, cb1, nullptr, h1, 204800, 32, 128, 32);

  // 2) conv2 (M=41472, N=64, K=512)
  im2col2_k<<<(41472 * 512 + 255) / 256, 256, 0, stream>>>(h1, imcolA);
  wmma_gemm_k<true, true, false, true><<<324, 256, 0, stream>>>(
      imcolA, cw2_16, cb2, nullptr, h2, 41472, 64, 512, 64);

  // 3) conv3 (M=25088, N=64, K=576)
  im2col3_k<<<(25088 * 576 + 255) / 256, 256, 0, stream>>>(h2, imcolA);
  wmma_gemm_k<true, true, false, true><<<196, 256, 0, stream>>>(
      imcolA, cw3_16, cb3, nullptr, h3, 25088, 64, 576, 64);

  // 4) fc (M=512, N=512, K=3136), keep f32 + f16 copies of x
  wmma_gemm_k<true, true, true, true><<<32, 256, 0, stream>>>(
      h3, fcw16, fcb, x32, x16, 512, 512, 3136, 512);

  // 5) heads ot/ct
  heads_k<<<(512 * 42 + 255) / 256, 256, 0, stream>>>(x32, otw, otb, ctw, ctb,
                                                      otB, ctB);

  // 6) m1 (M=512, N=10592 padded, K=512), bias, no relu
  wmma_gemm_k<false, true, true, false><<<662, 256, 0, stream>>>(
      x16, m1w16, m1b, m21, nullptr, 512, 10592, 512, 10584);

  // 7) decoder stage 1
  up21_42_k<<<(512 * 42 * 24 * 42 + 255) / 256, 256, 0, stream>>>(m21, m42u);
  dec1_conv_k<<<512, 256, 0, stream>>>(m42u, c1w16, c1b, t_u);

  // 8) decoder stage 2 + fused tail
  up42_84_k<<<(512 * 84 * 24 * 84 + 255) / 256, 256, 0, stream>>>(t_u, u84);
  dec2_fused_k<<<512, 256, 0, stream>>>(u84, c2w16, c2b, c3w, c3b, otB, ctB,
                                        inp, outp);
}
